// Attention_455266533741
// MI455X (gfx1250) — compile-verified, matrix operand order checked by reading, not run on hardware
//
#include <hip/hip_runtime.h>
#include <hip/hip_bf16.h>

// ---- problem constants ----
#define BB    2
#define NN    2048
#define DIMM  2048
#define HH    16
#define DHH   128
#define HD    2048            // H*DH
#define F3    6144            // 3*H*DH
#define MTOK  4096            // B*N
#define SOFTCAPF 50.0f
#define SCALEF   0.08838834764831845f   // 128^-0.5

#define LDSPITCH 40           // bf16 elems per LDS tile row (80B = 20 banks, conflict-free)

typedef __attribute__((ext_vector_type(16))) __bf16 v16bf;
typedef __attribute__((ext_vector_type(8)))  float  v8f;

union FragU { v16bf v; unsigned int u[8]; };

// Load one 16x32 bf16 WMMA A/B^T fragment from a row-major [rows, stride]
// bf16 matrix in GLOBAL memory with contiguous K.
// Per-lane layout (ISA 7.12.2): lane L holds row (L&15); k-pairs
// {0..7} and {16..23} shifted by khi = 8*(L>=16).
__device__ __forceinline__ v16bf load_frag(const __bf16* __restrict__ base,
                                           int row0, long stride, int k0) {
  int lane = threadIdx.x & 31;
  const __bf16* p = base + (long)(row0 + (lane & 15)) * stride + k0 + ((lane >> 4) << 3);
  FragU f;
#pragma unroll
  for (int i = 0; i < 4; ++i) {
    f.u[i]     = *(const unsigned int*)(p + 2 * i);
    f.u[i + 4] = *(const unsigned int*)(p + 16 + 2 * i);
  }
  return f.v;
}

// Same fragment gather from an LDS tile [128][LDSPITCH] (k0 folded into tile).
__device__ __forceinline__ v16bf load_frag_lds(const __bf16* tile, int row0) {
  int lane = threadIdx.x & 31;
  const __bf16* p = tile + (row0 + (lane & 15)) * LDSPITCH + ((lane >> 4) << 3);
  FragU f;
#pragma unroll
  for (int i = 0; i < 4; ++i) {
    f.u[i]     = *(const unsigned int*)(p + 2 * i);
    f.u[i + 4] = *(const unsigned int*)(p + 16 + 2 * i);
  }
  return f.v;
}

__device__ __forceinline__ v8f wmma_bf16(v16bf a, v16bf b, v8f c) {
  // (neg_a, A, neg_b, B, c_mod, C, reuse_a, reuse_b)
  return __builtin_amdgcn_wmma_f32_16x16x32_bf16(false, a, false, b, (short)0, c, false, false);
}

// ---- CDNA5 async DMA: global -> LDS, 16B per lane, tracked by ASYNCcnt ----
__device__ __forceinline__ void async_b128(const __bf16* gp, __bf16* lp) {
  unsigned lds = (unsigned)(unsigned long long)lp;  // low dword of generic ptr = LDS offset
  asm volatile("global_load_async_to_lds_b128 %0, %1, off"
               :: "v"(lds), "v"(gp) : "memory");
}

// Stage one 128x32 bf16 tile (rows of `stride`-pitched global matrix, cols
// [k0,k0+32)) into LDS [128][LDSPITCH]. 512 16B chunks over 256 threads.
__device__ __forceinline__ void async_copy_tile(const __bf16* __restrict__ g,
                                                long stride, int k0, __bf16* l) {
  int tid = threadIdx.x;
#pragma unroll
  for (int c = 0; c < 2; ++c) {
    int chunk = tid * 2 + c;
    int row = chunk >> 2, part = chunk & 3;
    async_b128(g + (long)row * stride + k0 + part * 8, l + row * LDSPITCH + part * 8);
  }
}

// Double-buffered async-staged WMMA mainloop: block tile 128(M)x128(N),
// wave tile 64x32 (4x2 accumulators), K stepped by 32.
__device__ __forceinline__ void wmma_gemm_tile(const __bf16* __restrict__ A,
                                               const __bf16* __restrict__ BT,
                                               int m0blk, int n0blk, int K,
                                               __bf16 (*sA)[128][LDSPITCH],
                                               __bf16 (*sB)[128][LDSPITCH],
                                               v8f acc[4][2]) {
  int wave = threadIdx.x >> 5;
  int wm = wave & 1, wn = wave >> 1;
  const __bf16* Ag = A + (long)m0blk * K;
  const __bf16* Bg = BT + (long)n0blk * K;
  int nk = K / 32;
  async_copy_tile(Ag, K, 0, &sA[0][0][0]);
  async_copy_tile(Bg, K, 0, &sB[0][0][0]);
  for (int s = 0; s < nk; ++s) {
    int buf = s & 1;
    if (s + 1 < nk) {
      async_copy_tile(Ag, K, (s + 1) * 32, &sA[buf ^ 1][0][0]);
      async_copy_tile(Bg, K, (s + 1) * 32, &sB[buf ^ 1][0][0]);
      if (s + 2 < nk) {  // L2 prefetch two stages ahead
        int row = threadIdx.x >> 1;
        __builtin_prefetch((const void*)(Ag + (long)row * K + (s + 2) * 32), 0, 1);
        __builtin_prefetch((const void*)(Bg + (long)row * K + (s + 2) * 32), 0, 1);
      }
      asm volatile("s_wait_asynccnt 0x4" ::: "memory");  // stage s landed (in-order)
    } else {
      asm volatile("s_wait_asynccnt 0x0" ::: "memory");
    }
    __syncthreads();
    v16bf a[4], b[2];
#pragma unroll
    for (int i = 0; i < 4; ++i) a[i] = load_frag_lds(&sA[buf][0][0], wm * 64 + 16 * i);
#pragma unroll
    for (int j = 0; j < 2; ++j) b[j] = load_frag_lds(&sB[buf][0][0], wn * 32 + 16 * j);
#pragma unroll
    for (int i = 0; i < 4; ++i)
#pragma unroll
      for (int j = 0; j < 2; ++j) acc[i][j] = wmma_bf16(a[i], b[j], acc[i][j]);
    __syncthreads();
  }
}

// ---------------- elementwise fp32 -> bf16 ----------------
__global__ void cvt_bf16_kernel(const float* __restrict__ in, __bf16* __restrict__ out, long n) {
  long i = (long)blockIdx.x * blockDim.x + threadIdx.x;
  if (i < n) out[i] = (__bf16)in[i];
}

// ---------------- transpose fp32[R,C] -> bf16[C,R] ----------------
__global__ void transpose_bf16_kernel(const float* __restrict__ in, __bf16* __restrict__ out,
                                      int R, int C) {
  __shared__ float tile[32][33];
  int tx = threadIdx.x, ty = threadIdx.y;
  int c = blockIdx.x * 32 + tx;
  int r = blockIdx.y * 32 + ty;
  tile[ty][tx] = in[(long)r * C + c];
  __syncthreads();
  int orow = blockIdx.x * 32 + ty;
  int oc   = blockIdx.y * 32 + tx;
  out[(long)orow * R + oc] = (__bf16)tile[tx][ty];
}

// ---------------- gates = sigmoid(x @ w_gates), [MTOK, H] fp32 ----------------
__global__ void gates_kernel(const float* __restrict__ x, const float* __restrict__ wg,
                             float* __restrict__ gates) {
  int idx = blockIdx.x * blockDim.x + threadIdx.x;  // t*16 + h
  int t = idx >> 4, h = idx & 15;
  const float* xr = x + (long)t * DIMM;
  float acc = 0.f;
  for (int c = 0; c < DIMM; ++c) acc += xr[c] * wg[c * HH + h];
  gates[idx] = 1.f / (1.f + __expf(-acc));
}

// ---------------- QKV GEMM (async-LDS staged) ----------------
__global__ __launch_bounds__(256) void gemm_qkv_kernel(
    const __bf16* __restrict__ A, const __bf16* __restrict__ BT,
    __bf16* __restrict__ qbuf, __bf16* __restrict__ kbuf, __bf16* __restrict__ vTbuf) {
  __shared__ __bf16 sA[2][128][LDSPITCH];
  __shared__ __bf16 sB[2][128][LDSPITCH];
  int wave = threadIdx.x >> 5;
  int lane = threadIdx.x & 31;
  int wm = wave & 1, wn = wave >> 1;
  int m0 = blockIdx.y * 128 + wm * 64;
  int n0 = blockIdx.x * 128 + wn * 32;
  v8f acc[4][2] = {};
  wmma_gemm_tile(A, BT, blockIdx.y * 128, blockIdx.x * 128, DIMM, sA, sB, acc);
  int col = lane & 15;
  int rbase = (lane >> 4) << 3;
#pragma unroll
  for (int i = 0; i < 4; ++i) {
#pragma unroll
    for (int j = 0; j < 2; ++j) {
      int f0  = n0 + 16 * j;
      int sel = f0 >> 11;           // 0=q 1=k 2=v
      int rem = f0 & 2047;
      int h = rem >> 7, d0 = rem & 127;
#pragma unroll
      for (int r = 0; r < 8; ++r) {
        int t   = m0 + 16 * i + rbase + r;
        int bb  = t >> 11, tok = t & 2047;
        float v = acc[i][j][r];
        long bh = (long)(bb * HH + h);
        if (sel == 0)      qbuf[(bh * NN + tok) * DHH + d0 + col] = (__bf16)(v * SCALEF);
        else if (sel == 1) kbuf[(bh * NN + tok) * DHH + d0 + col] = (__bf16)v;
        else               vTbuf[(bh * DHH + d0 + col) * NN + tok] = (__bf16)v;
      }
    }
  }
}

// ---------------- flash attention: one wave = one (b,h, 16-row q block) ----------------
__global__ __launch_bounds__(256) void attn_kernel(
    const __bf16* __restrict__ qbuf, const __bf16* __restrict__ kbuf,
    const __bf16* __restrict__ vTbuf, const float* __restrict__ gates,
    __bf16* __restrict__ Y) {
  __shared__ __bf16 sP[8][16][34];          // per-wave P-tile for D->A transpose
  int wave = threadIdx.x >> 5;
  int lane = threadIdx.x & 31;
  int g  = blockIdx.x * 8 + wave;
  int qb = g & 127;                          // 128 q-blocks per (b,h)
  int bh = g >> 7;
  int h = bh & (HH - 1), b = bh >> 4;
  int i0 = qb * 16;
  const __bf16* qbase = qbuf + (long)bh * NN * DHH;
  const __bf16* kbase = kbuf + (long)bh * NN * DHH;
  const __bf16* vbase = vTbuf + (long)bh * DHH * NN;

  v16bf qf[4];
#pragma unroll
  for (int ks = 0; ks < 4; ++ks) qf[ks] = load_frag(qbase, i0, DHH, ks * 32);

  v8f o[8] = {};
  float mrun[8], lrun[8];
#pragma unroll
  for (int r = 0; r < 8; ++r) { mrun[r] = -1e30f; lrun[r] = 0.f; }

  int rbase = (lane >> 4) << 3;
  int col = lane & 15;
  int nblk = (i0 + 16 + 31) >> 5;            // causal: only blocks up to the diagonal
  for (int jb = 0; jb < nblk; ++jb) {
    int j0 = jb * 32;
    v8f s0 = {}, s1 = {};
#pragma unroll
    for (int ks = 0; ks < 4; ++ks) {
      v16bf k0f = load_frag(kbase, j0,      DHH, ks * 32);
      v16bf k1f = load_frag(kbase, j0 + 16, DHH, ks * 32);
      s0 = wmma_bf16(qf[ks], k0f, s0);
      s1 = wmma_bf16(qf[ks], k1f, s1);
    }
    float mnew[8];
#pragma unroll
    for (int r = 0; r < 8; ++r) {
      int row = i0 + rbase + r;
      float a  = SOFTCAPF * tanhf(s0[r] * (1.0f / SOFTCAPF));
      float c2 = SOFTCAPF * tanhf(s1[r] * (1.0f / SOFTCAPF));
      if (j0 + col      > row) a  = -3.0e38f;
      if (j0 + 16 + col > row) c2 = -3.0e38f;
      s0[r] = a; s1[r] = c2;
      float rm = fmaxf(a, c2);
#pragma unroll
      for (int msk = 1; msk < 16; msk <<= 1) rm = fmaxf(rm, __shfl_xor(rm, msk, 32));
      mnew[r] = fmaxf(mrun[r], rm);
    }
#pragma unroll
    for (int r = 0; r < 8; ++r) {
      float sc = __expf(mrun[r] - mnew[r]);
      mrun[r] = mnew[r];
      float p0 = __expf(s0[r] - mnew[r]);
      float p1 = __expf(s1[r] - mnew[r]);
      float rs = p0 + p1;
#pragma unroll
      for (int msk = 1; msk < 16; msk <<= 1) rs += __shfl_xor(rs, msk, 32);
      lrun[r] = lrun[r] * sc + rs;
#pragma unroll
      for (int dt = 0; dt < 8; ++dt) o[dt][r] *= sc;
      sP[wave][rbase + r][col]      = (__bf16)p0;
      sP[wave][rbase + r][16 + col] = (__bf16)p1;
    }
    __builtin_amdgcn_wave_barrier();
    FragU pf;
    {
      const __bf16* pp = &sP[wave][lane & 15][(lane >> 4) << 3];
#pragma unroll
      for (int i2 = 0; i2 < 4; ++i2) {
        pf.u[i2]     = *(const unsigned int*)(pp + 2 * i2);
        pf.u[i2 + 4] = *(const unsigned int*)(pp + 16 + 2 * i2);
      }
    }
#pragma unroll
    for (int dt = 0; dt < 8; ++dt) {
      v16bf vf = load_frag(vbase, dt * 16, NN, j0);
      o[dt] = wmma_bf16(pf.v, vf, o[dt]);
    }
    __builtin_amdgcn_wave_barrier();
  }
  float fac[8];
#pragma unroll
  for (int r = 0; r < 8; ++r) {
    int tok = i0 + rbase + r;
    float gt = gates[((long)(b * NN) + tok) * HH + h];
    fac[r] = gt / lrun[r];
  }
#pragma unroll
  for (int dt = 0; dt < 8; ++dt)
#pragma unroll
    for (int r = 0; r < 8; ++r) {
      long t = (long)(b * NN) + i0 + rbase + r;
      Y[t * HD + h * DHH + dt * 16 + col] = (__bf16)(o[dt][r] * fac[r]);
    }
}

// ---------------- out GEMM (async-LDS staged): Y x woutT^T -> f32 ----------------
__global__ __launch_bounds__(256) void gemm_out_kernel(
    const __bf16* __restrict__ A, const __bf16* __restrict__ BT, float* __restrict__ out) {
  __shared__ __bf16 sA[2][128][LDSPITCH];
  __shared__ __bf16 sB[2][128][LDSPITCH];
  int wave = threadIdx.x >> 5;
  int lane = threadIdx.x & 31;
  int wm = wave & 1, wn = wave >> 1;
  int m0 = blockIdx.y * 128 + wm * 64;
  int n0 = blockIdx.x * 128 + wn * 32;
  v8f acc[4][2] = {};
  wmma_gemm_tile(A, BT, blockIdx.y * 128, blockIdx.x * 128, HD, sA, sB, acc);
  int col = lane & 15;
  int rbase = (lane >> 4) << 3;
#pragma unroll
  for (int i = 0; i < 4; ++i)
#pragma unroll
    for (int j = 0; j < 2; ++j)
#pragma unroll
      for (int r = 0; r < 8; ++r) {
        long t = m0 + 16 * i + rbase + r;
        out[t * DIMM + n0 + 16 * j + col] = acc[i][j][r];
      }
}

extern "C" void kernel_launch(void* const* d_in, const int* in_sizes, int n_in,
                              void* d_out, int out_size, void* d_ws, size_t ws_size,
                              hipStream_t stream) {
  const float* x    = (const float*)d_in[0];
  const float* wqkv = (const float*)d_in[1];
  const float* wg   = (const float*)d_in[2];
  const float* wout = (const float*)d_in[3];
  float* out = (float*)d_out;

  char* ws = (char*)d_ws;
  // layout (bytes): xbf 16MB (reused as Y) | wqkvT 24MB | woutT 8MB | q 16MB | k 16MB | vT 16MB | gates 256KB
  __bf16* xbf   = (__bf16*)(ws);
  __bf16* wqkvT = (__bf16*)(ws + 16777216);
  __bf16* woutT = (__bf16*)(ws + 41943040);
  __bf16* qb    = (__bf16*)(ws + 50331648);
  __bf16* kb    = (__bf16*)(ws + 67108864);
  __bf16* vT    = (__bf16*)(ws + 83886080);
  float*  gts   = (float*)(ws + 100663296);

  cvt_bf16_kernel<<<(MTOK * (long)DIMM) / 256, 256, 0, stream>>>(x, xbf, (long)MTOK * DIMM);
  transpose_bf16_kernel<<<dim3(F3 / 32, DIMM / 32), dim3(32, 32), 0, stream>>>(wqkv, wqkvT, DIMM, F3);
  transpose_bf16_kernel<<<dim3(DIMM / 32, HD / 32), dim3(32, 32), 0, stream>>>(wout, woutT, HD, DIMM);
  gates_kernel<<<(MTOK * HH) / 256, 256, 0, stream>>>(x, wg, gts);
  gemm_qkv_kernel<<<dim3(F3 / 128, MTOK / 128), 256, 0, stream>>>(xbf, wqkvT, qb, kb, vT);
  attn_kernel<<<512, 256, 0, stream>>>(qb, kb, vT, gts, xbf /* Y reuses xbf */);
  gemm_out_kernel<<<dim3(DIMM / 128, MTOK / 128), 256, 0, stream>>>(xbf, woutT, out);
}
